// RelativeGlobalAttention_81587198755290
// MI455X (gfx1250) — matrix operand order repaired and checked by reading, not compiled, not run on hardware
//
#include <hip/hip_runtime.h>

// ---------------------------------------------------------------------------
// Relative Global Attention for MI455X (gfx1250, wave32, WMMA)
// B=2, S=2048, D=512, H=8, DEPTH=64, M=2048 (M==S -> E used directly)
// All WMMA operands are contiguous 32B/lane loads (weights pre-transposed,
// V stored head-split transposed), except the softmax-weight A operand which
// requires a genuine cross-lane relayout (wave-private LDS bounce).
// ---------------------------------------------------------------------------

#define S_LEN   2048
#define D_MODEL 512
#define NB      2
#define NH      8
#define DEPTH   64
#define OUT0    (NB * S_LEN * D_MODEL)   // 2,097,152 floats, then [B,H,S,S]

typedef _Float16 half_t;
typedef half_t h16v __attribute__((ext_vector_type(16)));
typedef half_t h8v  __attribute__((ext_vector_type(8)));
typedef float  f8v  __attribute__((ext_vector_type(8)));

union H16 { h16v v; h8v h[2]; half_t e[16]; };

__device__ __forceinline__ f8v wmma_f16(h16v a, h16v b, f8v c) {
  return __builtin_amdgcn_wmma_f32_16x16x32_f16(false, a, false, b, (short)0, c,
                                                false, false);
}

// wave-level LDS fence (our LDS slices are wave-private; no block barrier)
__device__ __forceinline__ void wave_lds_sync() {
  __builtin_amdgcn_wave_barrier();
  asm volatile("s_wait_dscnt 0" ::: "memory");
  __builtin_amdgcn_wave_barrier();
}

// ---------------------------------------------------------------------------
// f32 -> f16 elementwise convert (inputs, E table)
// ---------------------------------------------------------------------------
__global__ __launch_bounds__(256) void cvt_f16_kernel(const float* __restrict__ in,
                                                      half_t* __restrict__ out,
                                                      int n) {
  int i = blockIdx.x * 256 + threadIdx.x;
  if (i < n) out[i] = (half_t)in[i];
}

// f32 [512x512] -> f16 transposed [512x512]: out[n][k] = in[k][n]
__global__ __launch_bounds__(256) void cvt_f16_t_kernel(const float* __restrict__ in,
                                                        half_t* __restrict__ out) {
  int idx = blockIdx.x * 256 + threadIdx.x;     // 262144 elements
  int k = idx >> 9, n = idx & 511;
  out[(size_t)n * D_MODEL + k] = (half_t)in[idx];
}

// ---------------------------------------------------------------------------
// Generic f16 GEMM: x16[BS x 512] @ W (stored transposed, f16) + bias.
// Block = 256 thr (8 waves), wave tile 16(M) x 64(N).  No LDS, no barriers:
// both operands are contiguous-per-lane b128 loads, W is L2-resident.
// mode 0: f32 row-major out (final projection)
// mode 1: f16 head-split   out[b,h,s,d]   (Q, K)
// mode 2: f16 head-split T out[b,h,d,s]   (V, for the AV B-operand)
// ---------------------------------------------------------------------------
__global__ __launch_bounds__(256) void gemm16_kernel(
    const half_t* __restrict__ x16, const half_t* __restrict__ w16t,
    const float* __restrict__ bias, void* __restrict__ outp, int mode) {
  const int tid  = threadIdx.x;
  const int lane = tid & 31, wave = tid >> 5;
  const int m0 = blockIdx.x * 128 + wave * 16;
  const int n0 = blockIdx.y * 64;
  const int r  = lane & 15, hi = lane >> 4;

  f8v acc[4] = {{}, {}, {}, {}};
  // A: row m0+r, K-run {k0+hi*16..+15}
  const half_t* xrow = x16 + (size_t)(m0 + r) * D_MODEL;
  // B: lane n holds K-run of 16 halves from transposed-W row n
  const half_t* wrow[4];
#pragma unroll
  for (int s = 0; s < 4; s++)
    wrow[s] = w16t + (size_t)(n0 + s * 16 + r) * D_MODEL + hi * 16;

  for (int k0 = 0; k0 < D_MODEL; k0 += 32) {
    H16 A;
    A.h[0] = *(const h8v*)(xrow + k0 + hi * 16);
    A.h[1] = *(const h8v*)(xrow + k0 + hi * 16 + 8);
#pragma unroll
    for (int s = 0; s < 4; s++) {
      h16v Bm = *(const h16v*)(wrow[s] + k0);
      acc[s] = wmma_f16(A.v, Bm, acc[s]);
    }
  }

  if (mode == 0) {
    float* out = (float*)outp;
#pragma unroll
    for (int s = 0; s < 4; s++) {
      int n = n0 + s * 16 + r;
      float bv = bias[n];
#pragma unroll
      for (int p = 0; p < 8; p++) {
        int m = m0 + p + (hi << 3);
        out[(size_t)m * D_MODEL + n] = acc[s][p] + bv;
      }
    }
  } else {
    half_t* out = (half_t*)outp;
#pragma unroll
    for (int s = 0; s < 4; s++) {
      int n = n0 + s * 16 + r;
      float bv = bias[n];
      int hh = n >> 6, dd = n & 63;
#pragma unroll
      for (int p = 0; p < 8; p++) {
        int m = m0 + p + (hi << 3);
        int bb = m >> 11, ss = m & (S_LEN - 1);
        size_t oi = (mode == 1)
                        ? ((((size_t)bb * NH + hh) * S_LEN + ss) * DEPTH + dd)
                        : ((((size_t)bb * NH + hh) * DEPTH + dd) * S_LEN + ss);
        out[oi] = (half_t)(acc[s][p] + bv);
      }
    }
  }
}

// ---------------------------------------------------------------------------
// Pass 1: logits = (q k^T + Srel) * depth^-0.5, causal-masked; raw logits to
// d_out weight region; per-row online max/sumexp -> stats ws.
// One wave per 16-row q-tile.  K^T and the E-band B-operands are contiguous
// 32B/lane loads; Srel diagonal pick via ds_bpermute.
// ---------------------------------------------------------------------------
__global__ __launch_bounds__(128) void attn_logits_kernel(
    const half_t* __restrict__ qh, const half_t* __restrict__ kh,
    const half_t* __restrict__ e16, float* __restrict__ attnw,
    float* __restrict__ statsM, float* __restrict__ statsL) {
  const int lane = threadIdx.x & 31;
  const int wave = threadIdx.x >> 5;
  const int bh = blockIdx.y;
  const int qt = blockIdx.x * 4 + wave;
  const int l0 = qt * 16;
  const int r  = lane & 15, hi = lane >> 4;

  const half_t* qb = qh + (size_t)bh * S_LEN * DEPTH;
  const half_t* kb = kh + (size_t)bh * S_LEN * DEPTH;
  float* aw = attnw + (size_t)bh * S_LEN * S_LEN;

  // q A operands for the whole tile (K=0..31 and 32..63 of DEPTH)
  const half_t* qrow = qb + (size_t)(l0 + r) * DEPTH;
  H16 Aq0, Aq1;
  Aq0.h[0] = *(const h8v*)(qrow + hi * 16);
  Aq0.h[1] = *(const h8v*)(qrow + hi * 16 + 8);
  Aq1.h[0] = *(const h8v*)(qrow + 32 + hi * 16);
  Aq1.h[1] = *(const h8v*)(qrow + 32 + hi * 16 + 8);

  float rm[8], rs[8];
#pragma unroll
  for (int p = 0; p < 8; p++) { rm[p] = -3.0e38f; rs[p] = 0.0f; }

  for (int j0 = 0; j0 <= l0; j0 += 16) {
    // ---- QK^T: B[kk,n] = k[j0+n, kk] -> contiguous 16-half run per lane
    const half_t* krow = kb + (size_t)(j0 + r) * DEPTH + hi * 16;
    if (j0 + 16 <= l0)  // warm L0 for the next key tile (global_prefetch_b8)
      __builtin_prefetch(krow + 16 * DEPTH, 0, 3);
    f8v acc = {};
    acc = wmma_f16(Aq0.v, *(const h16v*)(krow), acc);
    acc = wmma_f16(Aq1.v, *(const h16v*)(krow + 32), acc);

    // ---- Srel band: T = q @ F^T over E rows [base-15, base+16]
    const int base = (S_LEN - 1) - l0 + j0;
    int er0 = base - 15 + r;
    er0 = er0 < 0 ? 0 : (er0 > S_LEN - 1 ? S_LEN - 1 : er0);
    int er1 = base + 1 + r;
    er1 = er1 < 0 ? 0 : (er1 > S_LEN - 1 ? S_LEN - 1 : er1);
    const half_t* f0 = e16 + (size_t)er0 * DEPTH + hi * 16;
    const half_t* f1 = e16 + (size_t)er1 * DEPTH + hi * 16;
    f8v t0 = {}, t1 = {};
    t0 = wmma_f16(Aq0.v, *(const h16v*)(f0), t0);
    t0 = wmma_f16(Aq1.v, *(const h16v*)(f0 + 32), t0);
    t1 = wmma_f16(Aq0.v, *(const h16v*)(f1), t1);
    t1 = wmma_f16(Aq1.v, *(const h16v*)(f1 + 32), t1);

#pragma unroll
    for (int p = 0; p < 8; p++) {
      // Srel[r,c] = T[r, c - r + 15]; row-dependent lane shift within half
      int t  = (lane & 15) - p + 15 - (hi << 3);
      int sl = (((t & 15) + (lane & 16)) << 2);  // byte lane index, same half
      int g0 = __builtin_amdgcn_ds_bpermute(sl, __builtin_bit_cast(int, t0[p]));
      int g1 = __builtin_amdgcn_ds_bpermute(sl, __builtin_bit_cast(int, t1[p]));
      float srel = (t < 16) ? __builtin_bit_cast(float, g0)
                            : __builtin_bit_cast(float, g1);

      int jcol = j0 + (lane & 15);
      int lrow = l0 + p + (hi << 3);
      float logit = (acc[p] + srel) * 0.125f;  // DEPTH^-0.5 = 1/8
      if (jcol > lrow) logit = -1.0e9f;        // causal mask
      aw[(size_t)lrow * S_LEN + jcol] = logit;

      // online softmax stats; each 16-lane half holds one row
      float v = logit;
      for (int o = 1; o < 16; o <<= 1) v = fmaxf(v, __shfl_xor(v, o, 32));
      float nm = fmaxf(rm[p], v);
      float ex = __expf(logit - nm);
      for (int o = 1; o < 16; o <<= 1) ex += __shfl_xor(ex, o, 32);
      rs[p] = rs[p] * __expf(rm[p] - nm) + ex;
      rm[p] = nm;
    }
  }

  if ((lane & 15) == 0) {
#pragma unroll
    for (int p = 0; p < 8; p++) {
      int lrow = l0 + p + (hi << 3);
      statsM[(size_t)bh * S_LEN + lrow] = rm[p];
      statsL[(size_t)bh * S_LEN + lrow] = rs[p];
    }
  }
}

// ---------------------------------------------------------------------------
// Pass 2: normalize weights in place (zero the masked upper triangle), then
// attention = W @ V via WMMA.  Weights relayout C->A via wave-private LDS;
// V B-operand is a contiguous 32B/lane load from the transposed V buffer.
// ---------------------------------------------------------------------------
__global__ __launch_bounds__(128) void attn_av_kernel(
    float* __restrict__ attnw, const float* __restrict__ statsM,
    const float* __restrict__ statsL, const half_t* __restrict__ vT,
    half_t* __restrict__ concat16) {
  __shared__ __align__(16) half_t Aw[4][16][40];  // per-wave 16x32 +pad
  const int lane = threadIdx.x & 31;
  const int wave = threadIdx.x >> 5;
  const int bh = blockIdx.y;
  const int b = bh / NH, h = bh % NH;
  const int qt = blockIdx.x * 4 + wave;
  const int l0 = qt * 16;
  const int r  = lane & 15, hi = lane >> 4;

  float* aw = attnw + (size_t)bh * S_LEN * S_LEN;
  const half_t* vTb = vT + (size_t)bh * DEPTH * S_LEN;

  float rm[8], rinv[8];
#pragma unroll
  for (int p = 0; p < 8; p++) {
    int lrow = l0 + p + (hi << 3);
    rm[p]   = statsM[(size_t)bh * S_LEN + lrow];
    rinv[p] = 1.0f / statsL[(size_t)bh * S_LEN + lrow];
  }

  // V B-operand base pointers: lane holds d-column sv*16+r, K-run over keys
  const half_t* vcol[4];
#pragma unroll
  for (int sv = 0; sv < 4; sv++)
    vcol[sv] = vTb + (size_t)(sv * 16 + r) * S_LEN + hi * 16;

  f8v accv[4] = {{}, {}, {}, {}};

  for (int jt = 0; jt <= qt; jt += 2) {
#pragma unroll
    for (int sub = 0; sub < 2; sub++) {
      int jts = jt + sub;
      if (jts <= qt) {
#pragma unroll
        for (int p = 0; p < 8; p++) {
          int lrow = l0 + p + (hi << 3);
          size_t idx = (size_t)lrow * S_LEN + jts * 16 + r;
          float x = aw[idx];
          float w = __expf(x - rm[p]) * rinv[p];
          aw[idx] = w;                               // final weights out
          Aw[wave][p + (hi << 3)][sub * 16 + r] = (half_t)w;
        }
      } else {
#pragma unroll
        for (int p = 0; p < 8; p++)
          Aw[wave][p + (hi << 3)][sub * 16 + r] = (half_t)0.0f;
      }
    }
    wave_lds_sync();
    H16 A;
    A.h[0] = *(const h8v*)(&Aw[wave][r][hi * 16]);
    A.h[1] = *(const h8v*)(&Aw[wave][r][hi * 16 + 8]);
    wave_lds_sync();  // reads done before next iteration's stores

#pragma unroll
    for (int sv = 0; sv < 4; sv++) {
      h16v Bv = *(const h16v*)(vcol[sv] + jt * 16);  // keys jt*16.. contiguous
      accv[sv] = wmma_f16(A.v, Bv, accv[sv]);
    }
  }

  // zero the strictly-masked weight tiles (softmax -> exactly 0 there)
  for (int jt = qt + 1; jt < S_LEN / 16; jt++) {
#pragma unroll
    for (int p = 0; p < 8; p++) {
      int lrow = l0 + p + (hi << 3);
      aw[(size_t)lrow * S_LEN + jt * 16 + r] = 0.0f;
    }
  }

  // concat[b, s, h*64 + d] (f16)
#pragma unroll
  for (int sv = 0; sv < 4; sv++) {
#pragma unroll
    for (int p = 0; p < 8; p++) {
      int lrow = l0 + p + (hi << 3);
      size_t oi = ((size_t)b * S_LEN + lrow) * D_MODEL + h * DEPTH + sv * 16 + r;
      concat16[oi] = (half_t)accv[sv][p];
    }
  }
}

// ---------------------------------------------------------------------------
// Host launch
// ---------------------------------------------------------------------------
extern "C" void kernel_launch(void* const* d_in, const int* in_sizes, int n_in,
                              void* d_out, int out_size, void* d_ws,
                              size_t ws_size, hipStream_t stream) {
  (void)in_sizes; (void)n_in; (void)out_size; (void)ws_size;
  const float* v_in = (const float*)d_in[0];
  const float* k_in = (const float*)d_in[1];
  const float* q_in = (const float*)d_in[2];
  // d_in[3] = mask (bool) — causal mask is applied analytically
  const float* Wq   = (const float*)d_in[4];
  const float* Wq_b = (const float*)d_in[5];
  const float* Wk   = (const float*)d_in[6];
  const float* Wk_b = (const float*)d_in[7];
  const float* Wv   = (const float*)d_in[8];
  const float* Wv_b = (const float*)d_in[9];
  const float* Wo   = (const float*)d_in[10];
  const float* Wo_b = (const float*)d_in[11];
  const float* E    = (const float*)d_in[12];
  float* out = (float*)d_out;

  char* ws = (char*)d_ws;
  size_t off = 0;
  auto walloc = [&](size_t bytes) -> void* {
    void* p = ws + off;
    off += (bytes + 255) & ~(size_t)255;
    return p;
  };
  const size_t nW  = (size_t)D_MODEL * D_MODEL;           // 262144
  const size_t nE  = (size_t)S_LEN * DEPTH;               // 131072
  const size_t nX  = (size_t)NB * S_LEN * D_MODEL;        // 2097152
  const size_t nQH = (size_t)NB * NH * S_LEN * DEPTH;     // 2097152
  const size_t nST = (size_t)NB * NH * S_LEN;             // 32768

  half_t* Wqt = (half_t*)walloc(nW * 2);   // transposed f16 weights
  half_t* Wkt = (half_t*)walloc(nW * 2);
  half_t* Wvt = (half_t*)walloc(nW * 2);
  half_t* Wot = (half_t*)walloc(nW * 2);
  half_t* E16 = (half_t*)walloc(nE * 2);
  half_t* x16q = (half_t*)walloc(nX * 2);  // f16 activations
  half_t* x16k = (half_t*)walloc(nX * 2);
  half_t* x16v = (half_t*)walloc(nX * 2);
  half_t* qhp = (half_t*)walloc(nQH * 2);  // head-split q, k
  half_t* khp = (half_t*)walloc(nQH * 2);
  half_t* vTp = (half_t*)walloc(nQH * 2 + 256);  // head-split transposed V (+pad)
  half_t* c16 = (half_t*)walloc(nX * 2);
  float*  stM = (float*)walloc(nST * 4);
  float*  stL = (float*)walloc(nST * 4);

  // 1) weight transpose-convert, activation + E convert to f16
  cvt_f16_t_kernel<<<1024, 256, 0, stream>>>(Wq, Wqt);
  cvt_f16_t_kernel<<<1024, 256, 0, stream>>>(Wk, Wkt);
  cvt_f16_t_kernel<<<1024, 256, 0, stream>>>(Wv, Wvt);
  cvt_f16_t_kernel<<<1024, 256, 0, stream>>>(Wo, Wot);
  cvt_f16_kernel<<<(int)((nE + 255) / 256), 256, 0, stream>>>(E, E16, (int)nE);
  cvt_f16_kernel<<<(int)((nX + 255) / 256), 256, 0, stream>>>(q_in, x16q, (int)nX);
  cvt_f16_kernel<<<(int)((nX + 255) / 256), 256, 0, stream>>>(k_in, x16k, (int)nX);
  cvt_f16_kernel<<<(int)((nX + 255) / 256), 256, 0, stream>>>(v_in, x16v, (int)nX);

  // 2) projections (M=4096 -> 32 blocks of 128 rows; N=512 -> 8 of 64)
  dim3 gg(32, 8);
  gemm16_kernel<<<gg, 256, 0, stream>>>(x16q, Wqt, Wq_b, qhp, 1);
  gemm16_kernel<<<gg, 256, 0, stream>>>(x16k, Wkt, Wk_b, khp, 1);
  gemm16_kernel<<<gg, 256, 0, stream>>>(x16v, Wvt, Wv_b, vTp, 2);

  // 3) logits + Srel + causal mask + online softmax stats
  float* attnw = out + OUT0;
  dim3 ag(32, NB * NH);  // 4 waves/block * 32 blocks = 128 q-tiles per (b,h)
  attn_logits_kernel<<<ag, 128, 0, stream>>>(qhp, khp, E16, attnw, stM, stL);

  // 4) normalize weights in place + attention @ V -> concat16
  attn_av_kernel<<<ag, 128, 0, stream>>>(attnw, stM, stL, vTp, c16);

  // 5) output projection -> d_out[0 : B*S*D]
  gemm16_kernel<<<gg, 256, 0, stream>>>(c16, Wot, Wo_b, out, 0);
}
